// GAT_35235911696472
// MI455X (gfx1250) — matrix operand order repaired and checked by reading, not compiled, not run on hardware
//
#include <hip/hip_runtime.h>
#include <hip/hip_bf16.h>

#define N_NODES 65536
#define E_EDGES 1048576
#define ETOT    (E_EDGES + N_NODES)
#define HID     64
#define BN_EPS  1e-5f

typedef __attribute__((ext_vector_type(16))) _Float16 v16h;
typedef __attribute__((ext_vector_type(8)))  float    v8f;

// ---------- GEMM: [N x 64] @ [64 x 64] via v_wmma_f32_16x16x32_f16 ----------
// Block = 256 threads = 8 waves; each wave computes a 16(M) x 64(N) strip.
// W is transposed into LDS as f16 (Wt[n][k]) once per block.
__global__ __launch_bounds__(256)
void gemm64(const float* __restrict__ A, const float* __restrict__ W,
            const float* __restrict__ bias, float* __restrict__ Out, int relu_mode)
{
    __shared__ _Float16 Wt[64 * 72];   // row stride 72 f16 to spread banks

    const int tid = threadIdx.x;
    #pragma unroll
    for (int r = 0; r < 16; ++r) {
        int idx = tid + 256 * r;           // coalesced over W (row-major [k][j])
        int k = idx >> 6, j = idx & 63;
        Wt[j * 72 + k] = (_Float16)W[idx];
    }
    __syncthreads();

    const int wave = tid >> 5;
    const int lane = tid & 31;
    const int l    = lane & 15;
    const int hh   = lane >> 4;            // half-wave select
    const int rowbase = (blockIdx.x * 8 + wave) * 16;
    const int arow    = rowbase + l;

    v8f zero = {0.f,0.f,0.f,0.f,0.f,0.f,0.f,0.f};
    v8f acc[4] = {zero, zero, zero, zero};

    #pragma unroll
    for (int kc = 0; kc < 2; ++kc) {
        const int k0 = kc * 32;
        // A fragment: 16-bit A 16x32 layout -> lane reads K = [k0+hh*8, +8) and [k0+16+hh*8, +8)
        const float* Ap = A + (size_t)arow * HID + k0 + hh * 8;
        const float4* Ap4 = (const float4*)Ap;
        float4 a0 = Ap4[0], a1 = Ap4[1];   // K offsets 0..7
        float4 a2 = Ap4[4], a3 = Ap4[5];   // K offsets 16..23
        v16h af;
        af[0]=(_Float16)a0.x; af[1]=(_Float16)a0.y; af[2]=(_Float16)a0.z; af[3]=(_Float16)a0.w;
        af[4]=(_Float16)a1.x; af[5]=(_Float16)a1.y; af[6]=(_Float16)a1.z; af[7]=(_Float16)a1.w;
        af[8]=(_Float16)a2.x; af[9]=(_Float16)a2.y; af[10]=(_Float16)a2.z; af[11]=(_Float16)a2.w;
        af[12]=(_Float16)a3.x; af[13]=(_Float16)a3.y; af[14]=(_Float16)a3.z; af[15]=(_Float16)a3.w;

        #pragma unroll
        for (int c = 0; c < 4; ++c) {
            // B fragment: 32x16 f16, lane reads K = [k0+hh*8, +8) and [k0+16+hh*8, +8)
            const _Float16* Bp = &Wt[(c * 16 + l) * 72 + k0 + hh * 8];
            v16h bf;
            #pragma unroll
            for (int i = 0; i < 16; ++i) bf[i] = Bp[i < 8 ? i : i + 8];
            acc[c] = __builtin_amdgcn_wmma_f32_16x16x32_f16(
                false, af, false, bf, (short)0, acc[c], false, false);
        }
    }

    #pragma unroll
    for (int c = 0; c < 4; ++c) {
        const int col = c * 16 + l;
        const float b = relu_mode ? bias[col] : 0.f;
        #pragma unroll
        for (int j = 0; j < 8; ++j) {
            const int row = rowbase + hh * 8 + j;
            float v = acc[c][j] + b;
            if (relu_mode) v = fmaxf(v, 0.f);
            Out[(size_t)row * HID + col] = v;
        }
    }
}

// ---------- per-node attention dot products ----------
__global__ __launch_bounds__(256)
void att_dots(const float* __restrict__ g, const float* __restrict__ att_s,
              const float* __restrict__ att_d, float* __restrict__ asrc,
              float* __restrict__ adst)
{
    const int tid  = threadIdx.x;
    const int lane = tid & 31;
    const int n    = blockIdx.x * 8 + (tid >> 5);
    const float* gr = g + (size_t)n * HID + lane * 2;
    const float g0 = gr[0], g1 = gr[1];
    const int c = lane * 2;
    float s = g0 * att_s[c] + g1 * att_s[c + 1];
    float d = g0 * att_d[c] + g1 * att_d[c + 1];
    #pragma unroll
    for (int m = 16; m; m >>= 1) {
        s += __shfl_xor(s, m, 32);
        d += __shfl_xor(d, m, 32);
    }
    if (lane == 0) { asrc[n] = s; adst[n] = d; }
}

// ================= CSR build (once per launch; int atomics only) =================

__global__ void zero_deg(int* __restrict__ deg)
{
    deg[blockIdx.x * 256 + threadIdx.x] = 0;
}

__global__ void deg_hist(const int* __restrict__ dst, int* __restrict__ deg)
{
    const int e = blockIdx.x * 256 + threadIdx.x;
    if (e >= ETOT) return;
    const int d = (e < E_EDGES) ? dst[e] : (e - E_EDGES);
    atomicAdd(&deg[d], 1);
}

// per-block sums of deg (256 blocks x 256 elements)
__global__ __launch_bounds__(256)
void scan_block_sums(const int* __restrict__ deg, int* __restrict__ bsum)
{
    __shared__ int sh[256];
    const int t = threadIdx.x;
    sh[t] = deg[blockIdx.x * 256 + t];
    __syncthreads();
    for (int s = 128; s; s >>= 1) {
        if (t < s) sh[t] += sh[t + s];
        __syncthreads();
    }
    if (t == 0) bsum[blockIdx.x] = sh[0];
}

// exclusive scan of the 256 block sums (single block)
__global__ __launch_bounds__(256)
void scan_offsets(int* __restrict__ bsum)
{
    __shared__ int sh[256];
    const int t = threadIdx.x;
    sh[t] = bsum[t];
    __syncthreads();
    for (int o = 1; o < 256; o <<= 1) {
        int v = (t >= o) ? sh[t - o] : 0;
        __syncthreads();
        sh[t] += v;
        __syncthreads();
    }
    bsum[t] = t ? sh[t - 1] : 0;
}

// local exclusive scan + block offset -> rowptr, cursor
__global__ __launch_bounds__(256)
void scan_final(const int* __restrict__ deg, const int* __restrict__ bsum,
                int* __restrict__ rowptr, int* __restrict__ cursor)
{
    __shared__ int sh[256];
    const int t = threadIdx.x;
    const int idx = blockIdx.x * 256 + t;
    const int dv = deg[idx];
    sh[t] = dv;
    __syncthreads();
    for (int o = 1; o < 256; o <<= 1) {
        int v = (t >= o) ? sh[t - o] : 0;
        __syncthreads();
        sh[t] += v;
        __syncthreads();
    }
    const int val = sh[t] - dv + bsum[blockIdx.x];   // exclusive prefix
    rowptr[idx] = val;
    cursor[idx] = val;
    if (idx == N_NODES - 1) rowptr[N_NODES] = val + dv;
}

__global__ void csr_fill(const int* __restrict__ src, const int* __restrict__ dst,
                         int* __restrict__ cursor, int* __restrict__ csrc)
{
    const int e = blockIdx.x * 256 + threadIdx.x;
    if (e >= ETOT) return;
    int s, d;
    if (e < E_EDGES) { s = src[e]; d = dst[e]; } else { s = d = e - E_EDGES; }
    const int pos = atomicAdd(&cursor[d], 1);
    csrc[pos] = s;
}

// canonicalize per-node edge order (deterministic fp summation + gather locality)
__global__ void csr_sort(const int* __restrict__ rowptr, int* __restrict__ csrc)
{
    const int n = blockIdx.x * 256 + threadIdx.x;
    if (n >= N_NODES) return;
    const int a = rowptr[n], b = rowptr[n + 1];
    for (int i = a + 1; i < b; ++i) {
        const int v = csrc[i];
        int j = i - 1;
        while (j >= a && csrc[j] > v) { csrc[j + 1] = csrc[j]; --j; }
        csrc[j + 1] = v;
    }
}

// ============ fused segment-softmax + aggregation, wave per node, no atomics ============
__global__ __launch_bounds__(256)
void gat_aggregate(const int* __restrict__ rowptr, const int* __restrict__ csrc,
                   const float* __restrict__ asrc, const float* __restrict__ adst,
                   const float* __restrict__ g, float* __restrict__ hn)
{
    const int tid  = threadIdx.x;
    const int lane = tid & 31;
    const int d    = blockIdx.x * 8 + (tid >> 5);
    const int start = rowptr[d], end = rowptr[d + 1];
    const float ad = adst[d];

    // pass 1: max logit (lane-strided, then wave max-reduce)
    float m = -3.4e38f;
    for (int i = start + lane; i < end; i += 32) {
        float x = asrc[csrc[i]] + ad;
        x = x > 0.f ? x : 0.2f * x;
        m = fmaxf(m, x);
    }
    #pragma unroll
    for (int k = 16; k; k >>= 1) m = fmaxf(m, __shfl_xor(m, k, 32));

    // pass 2: sum of exp
    float ssum = 0.f;
    for (int i = start + lane; i < end; i += 32) {
        float x = asrc[csrc[i]] + ad;
        x = x > 0.f ? x : 0.2f * x;
        ssum += expf(x - m);
    }
    #pragma unroll
    for (int k = 16; k; k >>= 1) ssum += __shfl_xor(ssum, k, 32);
    const float inv = 1.f / ssum;   // self-loop guarantees ssum > 0

    // pass 3: serial over edges; each iteration: coalesced 256B row of g, fma into regs
    const int c = lane * 2;
    float acc0 = 0.f, acc1 = 0.f;
    for (int i = start; i < end; ++i) {
        const int s = csrc[i];                         // uniform across wave
        float x = asrc[s] + ad;
        x = x > 0.f ? x : 0.2f * x;
        const float w = expf(x - m) * inv;
        const float* gr = g + (size_t)s * HID + c;
        acc0 += w * gr[0];
        acc1 += w * gr[1];
    }
    float* o = hn + (size_t)d * HID + c;
    o[0] = acc0;
    o[1] = acc1;
}

// ---------- BN statistics: deterministic two-stage partial reduction ----------
__global__ __launch_bounds__(256)
void bn_stats(const float* __restrict__ hn, float* __restrict__ partial)
{
    __shared__ float ss[256], sq[256];
    const int tid = threadIdx.x;
    const int c = tid & 63, rg = tid >> 6;
    float s = 0.f, q = 0.f;
    for (int r = blockIdx.x * 4 + rg; r < N_NODES; r += gridDim.x * 4) {
        const float v = hn[(size_t)r * HID + c];
        s += v; q += v * v;
    }
    ss[tid] = s; sq[tid] = q;
    __syncthreads();
    if (tid < 64) {
        s = ss[tid] + ss[tid + 64] + ss[tid + 128] + ss[tid + 192];
        q = sq[tid] + sq[tid + 64] + sq[tid + 128] + sq[tid + 192];
        partial[blockIdx.x * 128 + tid]      = s;
        partial[blockIdx.x * 128 + 64 + tid] = q;
    }
}

__global__ void bn_reduce(const float* __restrict__ partial, float* __restrict__ stats)
{
    const int t = threadIdx.x;   // 128 threads
    float acc = 0.f;
    for (int b = 0; b < 256; ++b) acc += partial[b * 128 + t];   // fixed order
    stats[t] = acc;
}

// ---------- BN apply + ReLU (conv bias cancels inside BN) ----------
__global__ void bn_apply(const float* __restrict__ hn, const float* __restrict__ stats,
                         const float* __restrict__ gamma, const float* __restrict__ beta,
                         float* __restrict__ hout)
{
    const size_t idx = (size_t)blockIdx.x * 256 + threadIdx.x;
    const int c = (int)(idx & 63);
    const float inv_n = 1.f / (float)N_NODES;
    const float mu  = stats[c] * inv_n;
    const float var = stats[64 + c] * inv_n - mu * mu;
    float v = (hn[idx] - mu) * rsqrtf(var + BN_EPS) * gamma[c] + beta[c];
    hout[idx] = fmaxf(v, 0.f);
}

// ---------- head: 64 -> 2 GEMV + softmax, wave per node ----------
__global__ __launch_bounds__(256)
void head_softmax(const float* __restrict__ h, const float* __restrict__ Wout,
                  const float* __restrict__ bout, float* __restrict__ out)
{
    const int tid  = threadIdx.x;
    const int lane = tid & 31;
    const int n    = blockIdx.x * 8 + (tid >> 5);
    const float* hr = h + (size_t)n * HID + lane * 2;
    const float h0 = hr[0], h1 = hr[1];
    const int k = lane * 2;
    float l0 = h0 * Wout[k * 2 + 0] + h1 * Wout[(k + 1) * 2 + 0];
    float l1 = h0 * Wout[k * 2 + 1] + h1 * Wout[(k + 1) * 2 + 1];
    #pragma unroll
    for (int m = 16; m; m >>= 1) {
        l0 += __shfl_xor(l0, m, 32);
        l1 += __shfl_xor(l1, m, 32);
    }
    if (lane == 0) {
        l0 += bout[0]; l1 += bout[1];
        const float mm = fmaxf(l0, l1);
        const float p0 = expf(l0 - mm), p1 = expf(l1 - mm);
        const float inv = 1.f / (p0 + p1);
        out[(size_t)n * 2]     = p0 * inv;
        out[(size_t)n * 2 + 1] = p1 * inv;
    }
}

// ---------- driver ----------
extern "C" void kernel_launch(void* const* d_in, const int* in_sizes, int n_in,
                              void* d_out, int out_size, void* d_ws, size_t ws_size,
                              hipStream_t stream)
{
    // inputs in setup_inputs() insertion order (recursively flattened)
    const float* x    = (const float*)d_in[0];         // [N,64]
    const int*   ei   = (const int*)d_in[1];           // [2,E] int32
    const int*   src  = ei;
    const int*   dstA = ei + E_EDGES;
    const float* W_in = (const float*)d_in[2];
    const float* b_in = (const float*)d_in[3];
    const float* W_out= (const float*)d_in[22];
    const float* b_out= (const float*)d_in[23];

    // workspace carve-up (~58 MB)
    char* ws = (char*)d_ws;
    const size_t FN = (size_t)N_NODES * HID * sizeof(float);  // 16 MB
    float* h    = (float*)(ws);
    float* g    = (float*)(ws + FN);
    float* hn   = (float*)(ws + 2 * FN);
    char*  p    = ws + 3 * FN;
    float* asrc = (float*)p;                 p += (size_t)N_NODES * 4;
    float* adst = (float*)p;                 p += (size_t)N_NODES * 4;
    int*   deg  = (int*)p;                   p += (size_t)N_NODES * 4;
    int*   rowptr = (int*)p;                 p += (size_t)(N_NODES + 64) * 4;
    int*   cursor = (int*)p;                 p += (size_t)N_NODES * 4;
    int*   csrc   = (int*)p;                 p += (size_t)ETOT * 4;
    int*   bsum   = (int*)p;                 p += 256 * 4;
    float* partial= (float*)p;               p += 256 * 128 * 4;
    float* stats  = (float*)p;               p += 128 * 4;

    const dim3 blk(256);
    const int EB = (ETOT + 255) / 256;       // 4352
    const int NB = N_NODES / 256;            // 256

    // ---- CSR(dst) build: deterministic final order via per-node sort ----
    zero_deg<<<NB, blk, 0, stream>>>(deg);
    deg_hist<<<EB, blk, 0, stream>>>(dstA, deg);
    scan_block_sums<<<NB, blk, 0, stream>>>(deg, bsum);
    scan_offsets<<<1, blk, 0, stream>>>(bsum);
    scan_final<<<NB, blk, 0, stream>>>(deg, bsum, rowptr, cursor);
    csr_fill<<<EB, blk, 0, stream>>>(src, dstA, cursor, csrc);
    csr_sort<<<NB, blk, 0, stream>>>(rowptr, csrc);

    // ---- input layer: h = relu(x @ W_in + b_in) ----
    gemm64<<<N_NODES / 128, blk, 0, stream>>>(x, W_in, b_in, h, 1);

    for (int L = 0; L < 3; ++L) {
        const float* Wl    = (const float*)d_in[4 + 6 * L + 0];
        const float* atts  = (const float*)d_in[4 + 6 * L + 1];
        const float* attd  = (const float*)d_in[4 + 6 * L + 2];
        const float* gamma = (const float*)d_in[4 + 6 * L + 4];
        const float* beta  = (const float*)d_in[4 + 6 * L + 5];

        gemm64<<<N_NODES / 128, blk, 0, stream>>>(h, Wl, nullptr, g, 0);
        att_dots<<<N_NODES / 8, blk, 0, stream>>>(g, atts, attd, asrc, adst);
        gat_aggregate<<<N_NODES / 8, blk, 0, stream>>>(rowptr, csrc, asrc, adst, g, hn);
        bn_stats<<<256, blk, 0, stream>>>(hn, partial);
        bn_reduce<<<1, dim3(128), 0, stream>>>(partial, stats);
        bn_apply<<<(N_NODES * HID) / 256, blk, 0, stream>>>(hn, stats, gamma, beta, h);
    }

    head_softmax<<<N_NODES / 8, blk, 0, stream>>>(h, W_out, b_out, (float*)d_out);
}
